// CMME_44178033607327
// MI455X (gfx1250) — compile-verified, matrix operand order checked by reading, not run on hardware
//
#include <hip/hip_runtime.h>
#include <hip/hip_bf16.h>

// ---------- WMMA types (CDNA5 gfx1250, wave32) ----------
typedef __attribute__((ext_vector_type(16))) __bf16 v16bf;
typedef __attribute__((ext_vector_type(8)))  __bf16 v8bf;
typedef __attribute__((ext_vector_type(8)))  float  v8f;
typedef __attribute__((ext_vector_type(4)))  unsigned int v4u;
typedef __attribute__((ext_vector_type(8)))  int v8i_t;
typedef __attribute__((ext_vector_type(4)))  int v4i_t;

union FragAB { v16bf v; __bf16 e[16]; v8bf h2[2]; };
union FragC  { v8f v; float e[8]; };
union Pack8  { v8bf v; __bf16 e[8]; };

__device__ __forceinline__ __bf16 f2bf(float f) { return (__bf16)f; }

__device__ __forceinline__ v8bf zero8() {
  Pack8 z;
#pragma unroll
  for (int i = 0; i < 8; ++i) z.e[i] = f2bf(0.0f);
  return z.v;
}

__device__ __forceinline__ v8f wmma_bf16(v16bf a, v16bf b, v8f c) {
  // D = A(16x32 bf16) * B(32x16 bf16) + C(16x16 f32)
  return __builtin_amdgcn_wmma_f32_16x16x32_bf16(false, a, false, b, (short)0, c,
                                                 false, false);
}

#define BN_INV 0.9999950000374997f  // 1/sqrt(1 + 1e-5)

// Fragment build from a K-contiguous bf16 row (two 16B loads per 32-wide chunk).
__device__ __forceinline__ v16bf frag_from_row(const __bf16* row, int h, int kofs) {
  FragAB f;
  f.h2[0] = *(const v8bf*)(row + kofs + h * 8);
  f.h2[1] = *(const v8bf*)(row + kofs + 16 + h * 8);
  return f.v;
}

// ---------- Tensor Data Mover: 2D bf16 tile -> LDS ----------
#if defined(__gfx1250__) && __has_builtin(__builtin_amdgcn_tensor_load_to_lds) && \
    __has_builtin(__builtin_amdgcn_s_wait_tensorcnt)
#define HAVE_TDM 1
#else
#define HAVE_TDM 0
#endif

#if HAVE_TDM
// Loads `rows` rows of `row_elems` bf16 (row stride `stride_elems` in memory)
// compactly into LDS at byte offset lds_addr. Caller: one wave issues, then
// s_wait_tensorcnt(0) and a workgroup barrier.
__device__ __forceinline__ void tdm_load_2d_bf16(unsigned lds_addr,
                                                 const void* gsrc,
                                                 unsigned row_elems,
                                                 unsigned rows,
                                                 unsigned stride_elems) {
  const unsigned long long ga = (unsigned long long)(uintptr_t)gsrc;
  // D# group0: count=1 | lds_addr | global_addr[56:0] | type=2
  v4u g0 = {1u, lds_addr, (unsigned)ga,
            (unsigned)((ga >> 32) & 0x01FFFFFFu) | 0x80000000u};
  // D# group1: data_size=2B; tensor_dim0=stride, tensor_dim1=rows;
  // tile_dim0=row_elems, tile_dim1=rows; tensor_dim0_stride=stride.
  v8i_t g1 = {(int)0x00010000u,
              (int)((stride_elems & 0xFFFFu) << 16),
              (int)((stride_elems >> 16) | ((rows & 0xFFFFu) << 16)),
              (int)((rows >> 16) | ((row_elems & 0xFFFFu) << 16)),
              (int)(rows & 0xFFFFu),
              (int)stride_elems, 0, 0};
  v4i_t z4 = {0, 0, 0, 0};
#if __clang_major__ >= 23
  v8i_t z8 = {0, 0, 0, 0, 0, 0, 0, 0};
  __builtin_amdgcn_tensor_load_to_lds(g0, g1, z4, z4, z8, 0);
#else
  __builtin_amdgcn_tensor_load_to_lds(g0, g1, z4, z4, 0);
#endif
}
#endif

// ============================================================================
// One-time 3x3 weight repack: f32 [OC][IC][3][3] -> bf16 [tap][OC][IC]
// (coalesced writes; makes conv weight staging a contiguous 16B load).
// ============================================================================
__global__ __launch_bounds__(256) void k_repack_w3(
    const float* __restrict__ W, __bf16* __restrict__ Wb, int OC, int IC) {
  const int plane = OC * IC;
  const int total = plane * 9;
  for (int o = blockIdx.x * 256 + threadIdx.x; o < total;
       o += gridDim.x * 256) {
    const int tap = o / plane;
    const int r = o - tap * plane;  // oc*IC + ic
    Wb[o] = f2bf(W[(size_t)r * 9 + tap]);
  }
}

// ============================================================================
// Generic dilated 3x3 conv + BN + PReLU as a 9-tap GEMM.
// X: [B][IC][1024], Wb: bf16 [tap][OC][IC], Y: [B][OC][1024]
// Block: 256 threads (8 waves). Tile: 32 OC x 64 N, K-step 64 (2 WMMA/wave).
// Callers guarantee OC % 32 == 0 and IC % 64 == 0.
// ============================================================================
__global__ __launch_bounds__(256) void k_conv3(
    const float* __restrict__ X, const __bf16* __restrict__ Wb,
    const float* __restrict__ bias, const float* __restrict__ bn_g,
    const float* __restrict__ bn_beta, const float* __restrict__ alpha,
    float* __restrict__ Y, int IC, int OC, int dil) {
  const int N = 1024;
  __shared__ __align__(16) __bf16 Wl[32][72];
  __shared__ __align__(16) __bf16 Xl[64][72];
  const int b = blockIdx.z, oc0 = blockIdx.y * 32, n0 = blockIdx.x * 64;
  const int tid = threadIdx.x, wave = tid >> 5, lane = tid & 31;
  const int laneM = lane & 15, h = lane >> 4;
  const int ocT = wave & 1, nT = wave >> 1;
  const float* Xb = X + (size_t)b * IC * N;

  const int wr_r = tid >> 3, wr_c = (tid & 7) * 8;       // W tile: 8 k each
  const int xt_n = tid & 63, xt_k = (tid >> 6) * 16;     // X tile: 16 k each

  FragC acc;
#pragma unroll
  for (int j = 0; j < 8; ++j) acc.e[j] = 0.0f;

  for (int tap = 0; tap < 9; ++tap) {
    const int ty = tap / 3 - 1, tx = tap % 3 - 1;
    // branchless boundary handling: safe address + 0/1 mask
    const int n = n0 + xt_n;
    const int oh = n >> 5, ow = n & 31;
    const int ih = oh + ty * dil, iw = ow + tx * dil;
    const bool inb = ((unsigned)ih < 32u) && ((unsigned)iw < 32u);
    const int nn_s = inb ? ((ih << 5) + iw) : 0;
    const float msk = inb ? 1.0f : 0.0f;
    const __bf16* wsrc =
        Wb + ((size_t)tap * OC + (oc0 + wr_r)) * IC + wr_c;

    for (int k0 = 0; k0 < IC; k0 += 64) {
      // stage weight tile [32 oc][64 k]: one contiguous 16B load, no cvt
      *(v8bf*)&Wl[wr_r][wr_c] = *(const v8bf*)(wsrc + k0);
      {  // stage shifted activation tile transposed: Xl[n][k]
        Pack8 t0, t1;
#pragma unroll
        for (int i2 = 0; i2 < 8; ++i2)
          t0.e[i2] = f2bf(Xb[(size_t)(k0 + xt_k + i2) * N + nn_s] * msk);
#pragma unroll
        for (int i2 = 0; i2 < 8; ++i2)
          t1.e[i2] = f2bf(Xb[(size_t)(k0 + xt_k + 8 + i2) * N + nn_s] * msk);
        *(v8bf*)&Xl[xt_n][xt_k] = t0.v;
        *(v8bf*)&Xl[xt_n][xt_k + 8] = t1.v;
      }
      __syncthreads();
      const __bf16* wrow = &Wl[ocT * 16 + laneM][0];
      const __bf16* xrow = &Xl[nT * 16 + laneM][0];
      acc.v = wmma_bf16(frag_from_row(wrow, h, 0), frag_from_row(xrow, h, 0), acc.v);
      acc.v = wmma_bf16(frag_from_row(wrow, h, 32), frag_from_row(xrow, h, 32), acc.v);
      __syncthreads();
    }
  }
  const float av = alpha[0];
#pragma unroll
  for (int j = 0; j < 8; ++j) {
    const int oc = oc0 + ocT * 16 + j + 8 * h;
    const int n = n0 + nT * 16 + laneM;
    float v = acc.e[j] + bias[oc];
    v = v * (bn_g[oc] * BN_INV) + bn_beta[oc];
    v = (v >= 0.0f) ? v : av * v;
    Y[((size_t)b * OC + oc) * N + n] = v;
  }
}

// ============================================================================
// 1x1 conv GEMM, K-step 64. X: [B][IC][1024], W: [OC][IC].
// mode 0: plain conv        -> Yf[b][out_coff+oc][n], channel stride out_cstride
// mode 1: conv+BN+PReLU     -> Yf, same layout as mode 0
// mode 2: plain conv        -> Yt (bf16, grouped transposed): [(b*G+g)][n][gc]
// mode 3: plain conv        -> Yt (bf16, standard layout [b][oc][n])
// IC % 64 == 0; OC may be < 32 (rows masked branchlessly).
// ============================================================================
__global__ __launch_bounds__(256) void k_gemm1x1(
    const float* __restrict__ X, const float* __restrict__ W,
    const float* __restrict__ bias, const float* __restrict__ bn_g,
    const float* __restrict__ bn_beta, const float* __restrict__ alpha,
    float* __restrict__ Yf, __bf16* __restrict__ Yt,
    int IC, int OC, int mode, int gc, int out_coff, int out_cstride) {
  const int N = 1024;
  __shared__ __align__(16) __bf16 Wl[32][72];
  __shared__ __align__(16) __bf16 Xl[64][72];
  const int b = blockIdx.z, oc0 = blockIdx.y * 32, n0 = blockIdx.x * 64;
  const int tid = threadIdx.x, wave = tid >> 5, lane = tid & 31;
  const int laneM = lane & 15, h = lane >> 4;
  const int ocT = wave & 1, nT = wave >> 1;
  const float* Xb = X + (size_t)b * IC * N;

  const int wr_r = tid >> 3, wr_c = (tid & 7) * 8;
  const int xt_n = tid & 63, xt_k = (tid >> 6) * 16;
  const int wr_oc = oc0 + wr_r;
  const int wr_ocs = (wr_oc < OC) ? wr_oc : (OC - 1);
  const float wr_msk = (wr_oc < OC) ? 1.0f : 0.0f;

  FragC acc;
#pragma unroll
  for (int j = 0; j < 8; ++j) acc.e[j] = 0.0f;

  for (int k0 = 0; k0 < IC; k0 += 64) {
    {
      const float* wp = W + (size_t)wr_ocs * IC + k0 + wr_c;
#pragma unroll
      for (int i2 = 0; i2 < 8; ++i2)
        Wl[wr_r][wr_c + i2] = f2bf(wp[i2] * wr_msk);
    }
    {
      Pack8 t0, t1;
#pragma unroll
      for (int i2 = 0; i2 < 8; ++i2)
        t0.e[i2] = f2bf(Xb[(size_t)(k0 + xt_k + i2) * N + n0 + xt_n]);
#pragma unroll
      for (int i2 = 0; i2 < 8; ++i2)
        t1.e[i2] = f2bf(Xb[(size_t)(k0 + xt_k + 8 + i2) * N + n0 + xt_n]);
      *(v8bf*)&Xl[xt_n][xt_k] = t0.v;
      *(v8bf*)&Xl[xt_n][xt_k + 8] = t1.v;
    }
    __syncthreads();
    const __bf16* wrow = &Wl[ocT * 16 + laneM][0];
    const __bf16* xrow = &Xl[nT * 16 + laneM][0];
    acc.v = wmma_bf16(frag_from_row(wrow, h, 0), frag_from_row(xrow, h, 0), acc.v);
    acc.v = wmma_bf16(frag_from_row(wrow, h, 32), frag_from_row(xrow, h, 32), acc.v);
    __syncthreads();
  }
  const float av = alpha ? alpha[0] : 0.0f;
#pragma unroll
  for (int j = 0; j < 8; ++j) {
    const int oc = oc0 + ocT * 16 + j + 8 * h;
    if (oc >= OC) continue;
    const int n = n0 + nT * 16 + laneM;
    float v = acc.e[j] + bias[oc];
    if (mode == 1) {
      v = v * (bn_g[oc] * BN_INV) + bn_beta[oc];
      v = (v >= 0.0f) ? v : av * v;
    }
    if (mode == 2) {
      const int G = OC / gc;
      const int g = oc / gc, cc = oc - g * gc;
      Yt[((size_t)(b * G + g) * N + n) * gc + cc] = f2bf(v);
    } else if (mode == 3) {
      Yt[((size_t)b * out_cstride + out_coff + oc) * N + n] = f2bf(v);
    } else {
      Yf[((size_t)b * out_cstride + out_coff + oc) * N + n] = v;
    }
  }
}

// ============================================================================
// Attention energy: E[s][n][m] = sum_c QT[s][n][c] * KT[s][m][c], C = 16 or 32.
// The full KT slice ([1024][C] bf16, 32-64 KB) is staged into LDS with one
// Tensor-Data-Mover load; every wave's B-fragments then come from ds_load_b128.
// grid = (64 n-tiles, S). Each wave owns a 128-wide m strip (8 WMMA, unrolled).
// ============================================================================
__global__ __launch_bounds__(256) void k_energy(
    const __bf16* __restrict__ QT, const __bf16* __restrict__ KT,
    float* __restrict__ E, int C, size_t sStride) {
  __shared__ __align__(16) __bf16 Kl[1024 * 32];
  const int s = blockIdx.y;
  const int n0 = blockIdx.x * 16;
  const int tid = threadIdx.x, wave = tid >> 5, lane = tid & 31;
  const int laneM = lane & 15, h = lane >> 4;
  const __bf16* Qs = QT + sStride * s;
  const __bf16* Ks = KT + sStride * s;

#if HAVE_TDM
  if (wave == 0) {
    tdm_load_2d_bf16((unsigned)(uintptr_t)&Kl[0], Ks, (unsigned)C, 1024u,
                     (unsigned)C);
    __builtin_amdgcn_s_wait_tensorcnt(0);
  }
#else
  for (int idx = tid * 8; idx < 1024 * C; idx += 256 * 8)
    *(v8bf*)&Kl[idx] = *(const v8bf*)(Ks + idx);
#endif
  __syncthreads();

  FragAB A;
  {
    const __bf16* qp = Qs + (size_t)(n0 + laneM) * C;
    A.h2[0] = *(const v8bf*)(qp + h * 8);
    A.h2[1] = (C == 32) ? *(const v8bf*)(qp + 16 + h * 8) : zero8();
  }
  float* Es = E + (size_t)s * 1024 * 1024;
#pragma unroll
  for (int mt = 0; mt < 8; ++mt) {
    const int m0 = wave * 128 + mt * 16;
    FragAB Bm;
    const __bf16* kp = &Kl[(m0 + laneM) * C];
    Bm.h2[0] = *(const v8bf*)(kp + h * 8);
    Bm.h2[1] = (C == 32) ? *(const v8bf*)(kp + 16 + h * 8) : zero8();
    FragC d;
#pragma unroll
    for (int j = 0; j < 8; ++j) d.e[j] = 0.0f;
    d.v = wmma_bf16(A.v, Bm.v, d.v);
#pragma unroll
    for (int j = 0; j < 8; ++j)
      Es[(size_t)(n0 + j + 8 * h) * 1024 + m0 + laneM] = d.e[j];
  }
}

// ============================================================================
// Row softmax over 1024 columns; result stored as bf16 for WMMA consumption.
// ============================================================================
__global__ __launch_bounds__(256) void k_softmax(
    const float* __restrict__ E, __bf16* __restrict__ A) {
  __shared__ float red[256];
  const int row = blockIdx.x, tid = threadIdx.x;
  const float* er = E + (size_t)row * 1024;
  float v[4];
  float m = -3.4e38f;
#pragma unroll
  for (int i = 0; i < 4; ++i) { v[i] = er[tid + i * 256]; m = fmaxf(m, v[i]); }
  red[tid] = m; __syncthreads();
  for (int st = 128; st > 0; st >>= 1) {
    if (tid < st) red[tid] = fmaxf(red[tid], red[tid + st]);
    __syncthreads();
  }
  m = red[0]; __syncthreads();
  float ssum = 0.0f;
#pragma unroll
  for (int i = 0; i < 4; ++i) { v[i] = __expf(v[i] - m); ssum += v[i]; }
  red[tid] = ssum; __syncthreads();
  for (int st = 128; st > 0; st >>= 1) {
    if (tid < st) red[tid] += red[tid + st];
    __syncthreads();
  }
  const float inv = 1.0f / red[0];
  __bf16* ar = A + (size_t)row * 1024;
#pragma unroll
  for (int i = 0; i < 4; ++i) ar[tid + i * 256] = f2bf(v[i] * inv);
}

// ============================================================================
// O = V (Cv x 1024, bf16) * A^T, epilogue OUT = RES + gamma*O.
// The block's 64-row x 1024-col bf16 attention strip (128 KB) is staged into
// LDS with ONE Tensor-Data-Mover load; V fragments are direct 16B global
// loads (bf16, K-contiguous). K loop unrolled 4x -> 8 WMMA per body.
// ============================================================================
__global__ __launch_bounds__(256) void k_attn_out(
    const __bf16* __restrict__ V, const __bf16* __restrict__ A,
    const float* __restrict__ RES, float* __restrict__ OUT,
    const float* __restrict__ gammap,
    int Cv, size_t vS, size_t rS, size_t oS) {
  __shared__ __align__(16) __bf16 At[64][1024];
  const int s = blockIdx.z;
  const int c0 = blockIdx.y * 32;
  const int n0 = blockIdx.x * 64;
  const int tid = threadIdx.x, wave = tid >> 5, lane = tid & 31;
  const int laneM = lane & 15, h = lane >> 4;
  const int ocT = wave & 1, nT = wave >> 1;
  (void)Cv;

  const __bf16* gsrc = A + ((size_t)s * 1024 + n0) * 1024;
#if HAVE_TDM
  if (wave == 0) {
    tdm_load_2d_bf16((unsigned)(uintptr_t)&At[0][0], gsrc, 1024u, 64u, 1024u);
    __builtin_amdgcn_s_wait_tensorcnt(0);
  }
#else
  for (int idx = tid; idx < 8192; idx += 256) {
    const int r = idx >> 7, c = (idx & 127) * 8;
    *(v8bf*)&At[r][c] = *(const v8bf*)(gsrc + (size_t)r * 1024 + c);
  }
#endif
  __syncthreads();

  const __bf16* vp = V + vS * s + (size_t)(c0 + ocT * 16 + laneM) * 1024;
  const __bf16* arow = &At[nT * 16 + laneM][0];

  FragC acc;
#pragma unroll
  for (int j = 0; j < 8; ++j) acc.e[j] = 0.0f;

#pragma unroll 4
  for (int m0 = 0; m0 < 1024; m0 += 64) {
    __builtin_prefetch(vp + m0 + 512, 0, 1);
    acc.v = wmma_bf16(frag_from_row(vp, h, m0), frag_from_row(arow, h, m0), acc.v);
    acc.v = wmma_bf16(frag_from_row(vp, h, m0 + 32), frag_from_row(arow, h, m0 + 32), acc.v);
  }
  const float gm = gammap[0];
#pragma unroll
  for (int j = 0; j < 8; ++j) {
    const int c = c0 + ocT * 16 + j + 8 * h;
    const int n = n0 + nT * 16 + laneM;
    const size_t ri = rS * s + (size_t)c * 1024 + n;
    const size_t oi = oS * s + (size_t)c * 1024 + n;
    OUT[oi] = RES[ri] + gm * acc.e[j];
  }
}

// ---------------- small helpers ----------------
__global__ __launch_bounds__(256) void k_pool(const float* __restrict__ X,
                                              float* __restrict__ P) {
  __shared__ float red[256];
  const int idx = blockIdx.x, tid = threadIdx.x;
  const float* xr = X + (size_t)idx * 1024;
  float sv = xr[tid] + xr[tid + 256] + xr[tid + 512] + xr[tid + 768];
  red[tid] = sv; __syncthreads();
  for (int st = 128; st > 0; st >>= 1) {
    if (tid < st) red[tid] += red[tid + st];
    __syncthreads();
  }
  if (tid == 0) P[idx] = red[0] * (1.0f / 1024.0f);
}

__global__ __launch_bounds__(256) void k_conv5(
    const float* __restrict__ P, const float* __restrict__ W,
    const float* __restrict__ bias, const float* __restrict__ bn_g,
    const float* __restrict__ bn_beta, const float* __restrict__ alpha,
    float* __restrict__ concat) {
  __shared__ float red[256];
  __shared__ float vshared;
  const int oc = blockIdx.x, b = blockIdx.y, tid = threadIdx.x;
  red[tid] = W[(size_t)oc * 256 + tid] * P[b * 256 + tid];
  __syncthreads();
  for (int st = 128; st > 0; st >>= 1) {
    if (tid < st) red[tid] += red[tid + st];
    __syncthreads();
  }
  if (tid == 0) {
    float v = red[0] + bias[oc];
    v = v * (bn_g[oc] * BN_INV) + bn_beta[oc];
    v = (v >= 0.0f) ? v : alpha[0] * v;
    vshared = v;
  }
  __syncthreads();
  const float v = vshared;
  float* dst = concat + ((size_t)b * 640 + 512 + oc) * 1024;
  for (int i = tid; i < 1024; i += 256) dst[i] = v;
}

__global__ __launch_bounds__(256) void k_copy(const float* __restrict__ S,
                                              float* __restrict__ D, int n) {
  for (int t = blockIdx.x * 256 + threadIdx.x; t < n; t += gridDim.x * 256)
    D[t] = S[t];
}

// ============================================================================
extern "C" void kernel_launch(void* const* d_in, const int* in_sizes, int n_in,
                              void* d_out, int out_size, void* d_ws,
                              size_t ws_size, hipStream_t stream) {
  (void)in_sizes; (void)n_in; (void)out_size; (void)ws_size;
  const float* fusion = (const float*)d_in[0];
  const float* in_rgb = (const float*)d_in[1];
  const float* in_dsm = (const float*)d_in[2];
#define PF(i) ((const float*)d_in[i])
  float* out = (float*)d_out;

  char* ws = (char*)d_ws;
  size_t off = 0;
  auto alloc_f = [&](size_t elems) -> float* {
    float* p = (float*)(ws + off);
    off = (off + elems * 4 + 255) & ~(size_t)255;
    return p;
  };
  auto alloc_h = [&](size_t elems) -> __bf16* {
    __bf16* p = (__bf16*)(ws + off);
    off = (off + elems * 2 + 255) & ~(size_t)255;
    return p;
  };
  float* x   = alloc_f((size_t)4 * 256 * 1024);
  float* c2  = alloc_f((size_t)4 * 128 * 1024);
  float* c3  = alloc_f((size_t)4 * 128 * 1024);
  float* c4  = alloc_f((size_t)4 * 128 * 1024);
  __bf16* v2 = alloc_h((size_t)4 * 128 * 1024);
  __bf16* v3 = alloc_h((size_t)4 * 128 * 1024);
  __bf16* v4 = alloc_h((size_t)4 * 128 * 1024);
  __bf16* qT2 = alloc_h((size_t)4 * 1024 * 16);
  __bf16* kT2 = alloc_h((size_t)4 * 1024 * 16);
  __bf16* qT3 = alloc_h((size_t)4 * 1024 * 16);
  __bf16* kT3 = alloc_h((size_t)4 * 1024 * 16);
  __bf16* qT4 = alloc_h((size_t)4 * 1024 * 16);
  __bf16* kT4 = alloc_h((size_t)4 * 1024 * 16);
  float* concat = alloc_f((size_t)4 * 640 * 1024);
  float* fus    = alloc_f((size_t)4 * 256 * 1024);
  __bf16* qdT = alloc_h((size_t)32 * 1024 * 32);
  __bf16* kdT = alloc_h((size_t)32 * 1024 * 32);
  __bf16* qrT = alloc_h((size_t)32 * 1024 * 32);
  __bf16* krT = alloc_h((size_t)32 * 1024 * 32);
  __bf16* valb = alloc_h((size_t)4 * 256 * 1024);
  float* E    = alloc_f((size_t)4 * 1024 * 1024);
  __bf16* Ab  = alloc_h((size_t)4 * 1024 * 1024);
  float* pooled = alloc_f((size_t)4 * 256);
  __bf16* WbD = alloc_h((size_t)9 * 256 * 512);   // repacked down weights
  __bf16* Wb2 = alloc_h((size_t)9 * 128 * 256);
  __bf16* Wb3 = alloc_h((size_t)9 * 128 * 256);
  __bf16* Wb4 = alloc_h((size_t)9 * 128 * 256);

  const dim3 blk(256);
  const size_t S128 = (size_t)128 * 1024, S256 = (size_t)256 * 1024,
               S640 = (size_t)640 * 1024, SQK16 = (size_t)1024 * 16,
               SQK32 = (size_t)8 * 1024 * 32;

  // one-time 3x3 weight repacks (f32 OIHW -> bf16 [tap][OC][IC])
  k_repack_w3<<<dim3(1024), blk, 0, stream>>>(PF(3), WbD, 256, 512);
  k_repack_w3<<<dim3(512), blk, 0, stream>>>(PF(13), Wb2, 128, 256);
  k_repack_w3<<<dim3(512), blk, 0, stream>>>(PF(18), Wb3, 128, 256);
  k_repack_w3<<<dim3(512), blk, 0, stream>>>(PF(23), Wb4, 128, 256);

  // down: 3x3 512->256, dil 1
  k_conv3<<<dim3(16, 8, 4), blk, 0, stream>>>(fusion, WbD, PF(4), PF(5), PF(6), PF(7), x, 512, 256, 1);
  // conv1: 1x1 256->128 + BN+PReLU, directly into concat channels [0,128)
  k_gemm1x1<<<dim3(16, 4, 4), blk, 0, stream>>>(x, PF(8), PF(9), PF(10), PF(11), PF(12), concat, nullptr, 256, 128, 1, 0, 0, 640);
  // conv2/3/4: dilated 3x3 256->128
  k_conv3<<<dim3(16, 4, 4), blk, 0, stream>>>(x, Wb2, PF(14), PF(15), PF(16), PF(17), c2, 256, 128, 2);
  k_conv3<<<dim3(16, 4, 4), blk, 0, stream>>>(x, Wb3, PF(19), PF(20), PF(21), PF(22), c3, 256, 128, 4);
  k_conv3<<<dim3(16, 4, 4), blk, 0, stream>>>(x, Wb4, PF(24), PF(25), PF(26), PF(27), c4, 256, 128, 6);
  // q/k (bf16 transposed) and v (bf16 standard) projections per scale
  k_gemm1x1<<<dim3(16, 1, 4), blk, 0, stream>>>(c2, PF(38), PF(39), nullptr, nullptr, nullptr, nullptr, qT2, 128, 16, 2, 16, 0, 0);
  k_gemm1x1<<<dim3(16, 1, 4), blk, 0, stream>>>(c2, PF(40), PF(41), nullptr, nullptr, nullptr, nullptr, kT2, 128, 16, 2, 16, 0, 0);
  k_gemm1x1<<<dim3(16, 4, 4), blk, 0, stream>>>(c2, PF(42), PF(43), nullptr, nullptr, nullptr, nullptr, v2, 128, 128, 3, 0, 0, 128);
  k_gemm1x1<<<dim3(16, 1, 4), blk, 0, stream>>>(c3, PF(45), PF(46), nullptr, nullptr, nullptr, nullptr, qT3, 128, 16, 2, 16, 0, 0);
  k_gemm1x1<<<dim3(16, 1, 4), blk, 0, stream>>>(c3, PF(47), PF(48), nullptr, nullptr, nullptr, nullptr, kT3, 128, 16, 2, 16, 0, 0);
  k_gemm1x1<<<dim3(16, 4, 4), blk, 0, stream>>>(c3, PF(49), PF(50), nullptr, nullptr, nullptr, nullptr, v3, 128, 128, 3, 0, 0, 128);
  k_gemm1x1<<<dim3(16, 1, 4), blk, 0, stream>>>(c4, PF(52), PF(53), nullptr, nullptr, nullptr, nullptr, qT4, 128, 16, 2, 16, 0, 0);
  k_gemm1x1<<<dim3(16, 1, 4), blk, 0, stream>>>(c4, PF(54), PF(55), nullptr, nullptr, nullptr, nullptr, kT4, 128, 16, 2, 16, 0, 0);
  k_gemm1x1<<<dim3(16, 4, 4), blk, 0, stream>>>(c4, PF(56), PF(57), nullptr, nullptr, nullptr, nullptr, v4, 128, 128, 3, 0, 0, 128);
  // non-local out2 = gamma2*attn(q3,k2,v2) + c2 -> concat [128,256)
  k_energy<<<dim3(64, 4), blk, 0, stream>>>(qT3, kT2, E, 16, SQK16);
  k_softmax<<<dim3(4096), blk, 0, stream>>>(E, Ab);
  k_attn_out<<<dim3(16, 4, 4), blk, 0, stream>>>(v2, Ab, c2, concat + 128 * 1024, PF(44), 128, S128, S128, S640);
  // out3 = gamma3*attn(q4,k3,v3) + c3 -> concat [256,384)
  k_energy<<<dim3(64, 4), blk, 0, stream>>>(qT4, kT3, E, 16, SQK16);
  k_softmax<<<dim3(4096), blk, 0, stream>>>(E, Ab);
  k_attn_out<<<dim3(16, 4, 4), blk, 0, stream>>>(v3, Ab, c3, concat + 256 * 1024, PF(51), 128, S128, S128, S640);
  // out4 = gamma4*attn(q2,k4,v4) + c4 -> concat [384,512)
  k_energy<<<dim3(64, 4), blk, 0, stream>>>(qT2, kT4, E, 16, SQK16);
  k_softmax<<<dim3(4096), blk, 0, stream>>>(E, Ab);
  k_attn_out<<<dim3(16, 4, 4), blk, 0, stream>>>(v4, Ab, c4, concat + 384 * 1024, PF(58), 128, S128, S128, S640);
  // global pool + conv5 broadcast -> concat [512,640)
  k_pool<<<dim3(1024), blk, 0, stream>>>(x, pooled);
  k_conv5<<<dim3(128, 4), blk, 0, stream>>>(pooled, PF(28), PF(29), PF(30), PF(31), PF(32), concat);
  // fuse: 1x1 640->256 + BN+PReLU
  k_gemm1x1<<<dim3(16, 8, 4), blk, 0, stream>>>(concat, PF(33), PF(34), PF(35), PF(36), PF(37), fus, nullptr, 640, 256, 1, 0, 0, 256);
  // grouped projections (bf16 transposed, gc=32) + val (bf16 standard)
  k_gemm1x1<<<dim3(16, 8, 4), blk, 0, stream>>>(in_rgb, PF(59), PF(60), nullptr, nullptr, nullptr, nullptr, qrT, 256, 256, 2, 32, 0, 0);
  k_gemm1x1<<<dim3(16, 8, 4), blk, 0, stream>>>(in_rgb, PF(61), PF(62), nullptr, nullptr, nullptr, nullptr, krT, 256, 256, 2, 32, 0, 0);
  k_gemm1x1<<<dim3(16, 8, 4), blk, 0, stream>>>(in_dsm, PF(63), PF(64), nullptr, nullptr, nullptr, nullptr, qdT, 256, 256, 2, 32, 0, 0);
  k_gemm1x1<<<dim3(16, 8, 4), blk, 0, stream>>>(in_dsm, PF(65), PF(66), nullptr, nullptr, nullptr, nullptr, kdT, 256, 256, 2, 32, 0, 0);
  k_gemm1x1<<<dim3(16, 8, 4), blk, 0, stream>>>(fus, PF(67), PF(68), nullptr, nullptr, nullptr, nullptr, valb, 256, 256, 3, 0, 0, 256);
  // out = fus, then accumulate both grouped cross-attentions in place
  k_copy<<<dim3(1024), blk, 0, stream>>>(fus, out, 4 * 256 * 1024);
  for (int g = 0; g < 8; ++g) {  // dsm cross, gamma_dsm = PF(70)
    k_energy<<<dim3(64, 4), blk, 0, stream>>>(qdT + (size_t)g * 1024 * 32, kdT + (size_t)g * 1024 * 32, E, 32, SQK32);
    k_softmax<<<dim3(4096), blk, 0, stream>>>(E, Ab);
    k_attn_out<<<dim3(16, 1, 4), blk, 0, stream>>>(valb + (size_t)g * 32 * 1024, Ab, out + (size_t)g * 32 * 1024, out + (size_t)g * 32 * 1024, PF(70), 32, S256, S256, S256);
  }
  for (int g = 0; g < 8; ++g) {  // rgb cross, gamma_rgb = PF(69)
    k_energy<<<dim3(64, 4), blk, 0, stream>>>(qrT + (size_t)g * 1024 * 32, krT + (size_t)g * 1024 * 32, E, 32, SQK32);
    k_softmax<<<dim3(4096), blk, 0, stream>>>(E, Ab);
    k_attn_out<<<dim3(16, 1, 4), blk, 0, stream>>>(valb + (size_t)g * 32 * 1024, Ab, out + (size_t)g * 32 * 1024, out + (size_t)g * 32 * 1024, PF(69), 32, S256, S256, S256);
  }
#undef PF
}
